// GCN_36661840838723
// MI455X (gfx1250) — matrix operand order repaired and checked by reading, not compiled, not run on hardware
//
#include <hip/hip_runtime.h>
#include <hip/hip_bf16.h>

typedef __attribute__((ext_vector_type(16))) __bf16       v16bf;
typedef __attribute__((ext_vector_type(8)))  float        v8f;
typedef __attribute__((ext_vector_type(8)))  unsigned int v8u;

#define HID 128

union VU { v8u u; v16bf b; };

__device__ __forceinline__ unsigned short f2bf(float f) {
  unsigned u = __builtin_bit_cast(unsigned, f);
  u += 0x7FFFu + ((u >> 16) & 1u);          // round-to-nearest-even
  return (unsigned short)(u >> 16);
}
__device__ __forceinline__ unsigned packbf(float lo, float hi) {
  return ((unsigned)f2bf(hi) << 16) | (unsigned)f2bf(lo);
}

// ---------------- degree / norm ----------------
__global__ void k_deg_init(float* __restrict__ deg, int n) {
  int i = blockIdx.x * blockDim.x + threadIdx.x;
  if (i < n) deg[i] = 1.0f;                  // self-loop
}
__global__ void k_deg(const int* __restrict__ dst, float* __restrict__ deg, int E) {
  int e = blockIdx.x * blockDim.x + threadIdx.x;
  if (e < E) atomicAdd(&deg[dst[e]], 1.0f);
}
__global__ void k_dinv(float* __restrict__ deg, int n) {
  int i = blockIdx.x * blockDim.x + threadIdx.x;
  if (i < n) { float d = deg[i]; deg[i] = (d > 0.f) ? rsqrtf(d) : 0.f; }
}

// f32 -> bf16 mirror (for layer-0 input x)
__global__ void k_cvt_bf16(const float* __restrict__ in, unsigned short* __restrict__ out,
                           int total) {
  int idx = blockIdx.x * blockDim.x + threadIdx.x;
  if (idx < total) out[idx] = f2bf(in[idx]);
}

// ---------------- WMMA GEMM: T[M,128] = A[M,128](bf16) @ W[128,128](f32->bf16) --------
// 256 threads = 8 waves; each wave computes a 16x128 row strip with
// v_wmma_f32_16x16x32_bf16 (8 N-tiles x 4 K-steps = 32 WMMA / wave).
// A fragments load directly as two global b128 of contiguous bf16 per K-step.
__global__ __launch_bounds__(256) void k_gemm_bf16(
    const unsigned short* __restrict__ A, const float* __restrict__ W,
    float* __restrict__ T, int M) {
  // W staged in B-fragment-major order: frag(kt,j) = 32 lanes x 8 dwords.
  __shared__ unsigned ldsB[4 * 8 * 32 * 8];  // 32 KB
  const int tid = threadIdx.x;
  for (int idx = tid; idx < 8192; idx += 256) {
    int kt   = idx >> 11;          // K-step (32 wide)
    int j    = (idx >> 8) & 7;     // N tile
    int lane = (idx >> 3) & 31;
    int i    = idx & 7;            // operand VGPR
    int half = lane >> 4;
    int n    = j * 16 + (lane & 15);
    int k    = kt * 32 + half * 8 + (i >> 2) * 8 + 2 * i;  // B: VGPR0-3 K=half*8+{0..7}, VGPR4-7 K=16+half*8+{0..7}
    ldsB[idx] = packbf(W[k * HID + n], W[(k + 1) * HID + n]);
  }
  __syncthreads();

  const int wave = tid >> 5;
  const int lane = tid & 31;
  const int half = lane >> 4;
  const int rowBase = blockIdx.x * 128 + wave * 16;
  int mrow  = rowBase + (lane & 15);
  int mload = (mrow < M) ? mrow : (M - 1);   // clamp: keep EXEC all-ones
  const unsigned short* arow = A + (size_t)mload * HID;

  v8f acc[8] = {};
#pragma unroll
  for (int kt = 0; kt < 4; ++kt) {
    // A fragment: VGPR0-3 K = kb+half*8+{0..7}; VGPR4-7 K = kb+16+half*8+{0..7}
    // Contiguous bf16 in row-major memory -> two 16B loads, no conversion.
    int c0 = kt * 32 + half * 8;
    uint4 q0 = *(const uint4*)(arow + c0);
    uint4 q1 = *(const uint4*)(arow + c0 + 16);
    VU av;
    av.u[0] = q0.x; av.u[1] = q0.y; av.u[2] = q0.z; av.u[3] = q0.w;
    av.u[4] = q1.x; av.u[5] = q1.y; av.u[6] = q1.z; av.u[7] = q1.w;

    const unsigned* base = &ldsB[kt * 2048 + lane * 8];
#pragma unroll
    for (int j = 0; j < 8; ++j) {
      VU bv;
#pragma unroll
      for (int i = 0; i < 8; ++i) bv.u[i] = base[j * 256 + i];
      acc[j] = __builtin_amdgcn_wmma_f32_16x16x32_bf16(
          false, av.b, false, bv.b, (short)0, acc[j], false, false);
    }
  }

  // D layout: lane n = l&15, VGPR i -> row half*8 + i
  const int n0 = lane & 15;
#pragma unroll
  for (int j = 0; j < 8; ++j) {
#pragma unroll
    for (int i = 0; i < 8; ++i) {
      int r = rowBase + half * 8 + i;
      if (r < M) T[(size_t)r * HID + j * 16 + n0] = acc[j][i];
    }
  }
}

// ---------------- aggregation ----------------
// Q[i,:] = dinv[i]^2 * T[i,:]   (self-loop term, also initializes Q)
__global__ void k_selfloop(const float* __restrict__ T, const float* __restrict__ dinv,
                           float* __restrict__ Q, int total) {
  int idx = blockIdx.x * blockDim.x + threadIdx.x;
  if (idx < total) {
    float di = dinv[idx >> 7];
    Q[idx] = T[idx] * di * di;
  }
}
// Q[dst,:] += dinv[src]*dinv[dst] * T[src,:]   — one wave per edge
__global__ __launch_bounds__(256) void k_scatter(
    const float* __restrict__ T, const int* __restrict__ src,
    const int* __restrict__ dst, const float* __restrict__ dinv,
    float* __restrict__ Q, int E) {
  int e = blockIdx.x * 8 + (threadIdx.x >> 5);
  if (e >= E) return;
  int lane = threadIdx.x & 31;
  int s = src[e], d = dst[e];
  float w = dinv[s] * dinv[d];
  float4 v = *(const float4*)(T + (size_t)s * HID + lane * 4);
  float* o = Q + (size_t)d * HID + lane * 4;
  atomicAdd(o + 0, v.x * w);
  atomicAdd(o + 1, v.y * w);
  atomicAdd(o + 2, v.z * w);
  atomicAdd(o + 3, v.w * w);
}

// bias + ReLU in f32, plus bf16 mirror for the next layer's WMMA GEMM
__global__ void k_bias_relu_cvt(float* __restrict__ Q, const float* __restrict__ b,
                                unsigned short* __restrict__ Hb, int total) {
  int idx = blockIdx.x * blockDim.x + threadIdx.x;
  if (idx < total) {
    float v = Q[idx] + b[idx & (HID - 1)];
    v = v > 0.f ? v : 0.f;
    Q[idx]  = v;
    Hb[idx] = f2bf(v);
  }
}

// ---------------- pooling + head ----------------
__global__ void k_zero(float* p, int n) {
  int i = blockIdx.x * blockDim.x + threadIdx.x;
  if (i < n) p[i] = 0.f;
}
__global__ void k_pool(const float* __restrict__ h, float* __restrict__ pool, int N) {
  int f  = threadIdx.x;            // 0..127
  int n0 = blockIdx.x * 128;
  int lim = (n0 + 128 < N) ? (n0 + 128) : N;
  float s = 0.f;
  for (int n = n0; n < lim; ++n) s += h[(size_t)n * HID + f];
  atomicAdd(&pool[f], s);
}
__global__ void k_head(const float* __restrict__ pool, const float* __restrict__ fcw,
                       const float* __restrict__ fcb, float* __restrict__ out, float invN) {
  int o = threadIdx.x;
  if (o < 6) {
    float s = fcb[o];
    for (int f = 0; f < HID; ++f) s += pool[f] * invN * fcw[f * 6 + o];
    out[o] = s;
  }
}

extern "C" void kernel_launch(void* const* d_in, const int* in_sizes, int n_in,
                              void* d_out, int out_size, void* d_ws, size_t ws_size,
                              hipStream_t stream) {
  (void)n_in; (void)out_size; (void)ws_size;
  const float* x   = (const float*)d_in[0];
  const int*   ei  = (const int*)d_in[1];
  const float* Ws[3] = {(const float*)d_in[2], (const float*)d_in[4], (const float*)d_in[6]};
  const float* bs[3] = {(const float*)d_in[3], (const float*)d_in[5], (const float*)d_in[7]};
  const float* fcw = (const float*)d_in[8];
  const float* fcb = (const float*)d_in[9];
  float* out = (float*)d_out;

  const int N = in_sizes[0] / HID;     // 100000
  const int E = in_sizes[1] / 2;       // 1600000
  const int* srcIdx = ei;
  const int* dstIdx = ei + E;
  const int NE = N * HID;

  // workspace layout (floats):
  // [dinv(N,pad)] [P(N*128) f32] [Q(N*128) f32] [pool(128,pad)] [Hb(N*128) bf16]
  float* ws   = (float*)d_ws;
  size_t o0   = ((size_t)N + 511) & ~(size_t)511;
  float* dinv = ws;
  float* P    = ws + o0;
  float* Q    = P + (size_t)N * HID;
  float* pool = Q + (size_t)N * HID;
  unsigned short* Hb = (unsigned short*)(pool + 128);   // 16B-aligned (offsets are /512 floats)

  // symmetric GCN normalization
  k_deg_init<<<(N + 255) / 256, 256, 0, stream>>>(dinv, N);
  k_deg     <<<(E + 255) / 256, 256, 0, stream>>>(dstIdx, dinv, E);
  k_dinv    <<<(N + 255) / 256, 256, 0, stream>>>(dinv, N);

  // layer-0 input in bf16
  k_cvt_bf16<<<(NE + 255) / 256, 256, 0, stream>>>(x, Hb, NE);

  for (int l = 0; l < 3; ++l) {
    k_gemm_bf16    <<<(N + 127) / 128, 256, 0, stream>>>(Hb, Ws[l], P, N);
    k_selfloop     <<<(NE + 255) / 256, 256, 0, stream>>>(P, dinv, Q, NE);
    k_scatter      <<<(E + 7) / 8, 256, 0, stream>>>(P, srcIdx, dstIdx, dinv, Q, E);
    k_bias_relu_cvt<<<(NE + 255) / 256, 256, 0, stream>>>(Q, bs[l], Hb, NE);
  }

  k_zero<<<1, 128, 0, stream>>>(pool, HID);
  k_pool<<<(N + 127) / 128, 128, 0, stream>>>(Q, pool, N);
  k_head<<<1, 32, 0, stream>>>(pool, fcw, fcb, out, 1.0f / (float)N);
}
